// QLiteMSA_40166534152651
// MI455X (gfx1250) — 8-bit matrix operand order repaired and checked by reading, compiled, not run on hardware
//
#include <hip/hip_runtime.h>
#include <cstdint>
#include <cstddef>


typedef __attribute__((ext_vector_type(8))) int v8i;
typedef __attribute__((ext_vector_type(4))) int v4i;
typedef __attribute__((ext_vector_type(4))) unsigned int v4u;

#define BATCH   32
#define CIN     256
#define HW      1024
#define DIM     8
#define HEADS   32
#define TOTAL   256
#define H2      64
#define C2      512

#if defined(__has_builtin)
#  if __has_builtin(__builtin_amdgcn_tensor_load_to_lds) && \
      __has_builtin(__builtin_amdgcn_s_wait_tensorcnt)
#    define USE_TDM 1
#  endif
#endif
#ifndef USE_TDM
#  define USE_TDM 0
#endif

__device__ __forceinline__ float fq8(float v, float s) {
  return rintf(fminf(fmaxf(v / s, -128.f), 127.f));
}

__global__ void k_quant_input(const float* __restrict__ x,
                              const float* __restrict__ s_in,
                              int8_t* __restrict__ qx) {
  size_t idx = (size_t)blockIdx.x * 256 + threadIdx.x;
  int n = idx & (HW - 1);
  int c = (idx >> 10) & (CIN - 1);
  int b = (int)(idx >> 18);
  float s = *s_in;
  qx[((size_t)b * HW + n) * CIN + c] = (int8_t)(int)fq8(x[idx], s);
}

__global__ void k_quant_weight(const float* __restrict__ W,
                               const float* __restrict__ sW,
                               int8_t* __restrict__ qW,
                               int perCh, int total) {
  int idx = blockIdx.x * 256 + threadIdx.x;
  if (idx >= total) return;
  qW[idx] = (int8_t)(int)fq8(W[idx], sW[idx / perCh]);
}

__device__ __forceinline__ v8i load_a_frag(const int8_t* __restrict__ A,
                                           int K, int mBase, int kb, int lane) {
  int m = mBase + (lane & 15);
  const int8_t* p = A + (size_t)m * K + kb + ((lane >> 4) << 3);
  v8i a;
  a[0] = *(const int*)(p + 0);   a[1] = *(const int*)(p + 4);
  a[2] = *(const int*)(p + 16);  a[3] = *(const int*)(p + 20);
  a[4] = *(const int*)(p + 32);  a[5] = *(const int*)(p + 36);
  a[6] = *(const int*)(p + 48);  a[7] = *(const int*)(p + 52);
  return a;
}

__device__ __forceinline__ v8i load_b_frag(const int8_t* __restrict__ Bt,
                                           int K, int nBase, int kb, int lane) {
  int n = nBase + (lane & 15);
  int kh = (lane >> 4) << 3;
  const int8_t* p = Bt + (size_t)n * K + kb;
  v8i b;
  b[0] = *(const int*)(p + kh + 0);       b[1] = *(const int*)(p + kh + 4);
  b[2] = *(const int*)(p + kh + 16);      b[3] = *(const int*)(p + kh + 20);
  b[4] = *(const int*)(p + 32 + kh + 0);  b[5] = *(const int*)(p + 32 + kh + 4);
  b[6] = *(const int*)(p + 32 + kh + 16); b[7] = *(const int*)(p + 32 + kh + 20);
  return b;
}

__device__ __forceinline__ void gemm_tile_16x32(const int8_t* __restrict__ A,
                                                const int8_t* __restrict__ Bb,
                                                int K, int mBase, int nBase,
                                                int lane, v8i& c0, v8i& c1) {
  c0 = (v8i){0, 0, 0, 0, 0, 0, 0, 0};
  c1 = (v8i){0, 0, 0, 0, 0, 0, 0, 0};
  for (int kb = 0; kb < K; kb += 64) {
    v8i a  = load_a_frag(A, K, mBase, kb, lane);
    v8i b0 = load_b_frag(Bb, K, nBase, kb, lane);
    v8i b1 = load_b_frag(Bb, K, nBase + 16, kb, lane);
    if (kb + 64 < K) {
      __builtin_prefetch(A + (size_t)(mBase + (lane & 15)) * K + kb + 64, 0, 3);
      __builtin_prefetch(Bb + (size_t)(nBase + (lane & 15)) * K + kb + 64, 0, 3);
    }
    c0 = __builtin_amdgcn_wmma_i32_16x16x64_iu8(true, a, true, b0, c0, false, false);
    c1 = __builtin_amdgcn_wmma_i32_16x16x64_iu8(true, a, true, b1, c1, false, false);
  }
}

__global__ void k_gemm_iu8_quant(const int8_t* __restrict__ A,
                                 const int8_t* __restrict__ Bt,
                                 int8_t* __restrict__ Out,
                                 const float* __restrict__ sW,
                                 const float* __restrict__ sIn,
                                 const float* __restrict__ sOut,
                                 int M, int N, int K) {
  int lane = threadIdx.x & 31;
  int wave = threadIdx.x >> 5;
  int mBase = blockIdx.x * 32 + (wave >> 2) * 16;
  int nBase = blockIdx.y * 128 + (wave & 3) * 32;
  const int8_t* Bb = Bt + (size_t)blockIdx.z * N * K;
  int8_t* Ob = Out + (size_t)blockIdx.z * M * N;

  v8i c0, c1;
  gemm_tile_16x32(A, Bb, K, mBase, nBase, lane, c0, c1);

  float si = *sIn, so = *sOut;
  int n0 = nBase + (lane & 15);
  int mrow = mBase + ((lane >> 4) << 3);
  for (int r = 0; r < 8; ++r) {
    int m = mrow + r;
    float sc = si * sW[m];
    Ob[(size_t)m * N + n0]      = (int8_t)(int)fq8((float)c0[r] * sc, so);
    Ob[(size_t)m * N + n0 + 16] = (int8_t)(int)fq8((float)c1[r] * sc, so);
  }
}

__global__ void k_gemm_iu8_f32(const int8_t* __restrict__ A,
                               const int8_t* __restrict__ Bt,
                               float* __restrict__ Out,
                               const float* __restrict__ sW,
                               const float* __restrict__ sIn,
                               int M, int N, int K) {
  int lane = threadIdx.x & 31;
  int wave = threadIdx.x >> 5;
  int mBase = blockIdx.x * 32 + (wave >> 2) * 16;
  int nBase = blockIdx.y * 128 + (wave & 3) * 32;
  const int8_t* Bb = Bt + (size_t)blockIdx.z * N * K;
  float* Ob = Out + (size_t)blockIdx.z * M * N;

  v8i c0, c1;
  gemm_tile_16x32(A, Bb, K, mBase, nBase, lane, c0, c1);

  float si = *sIn;
  int n0 = nBase + (lane & 15);
  int mrow = mBase + ((lane >> 4) << 3);
  for (int r = 0; r < 8; ++r) {
    int m = mrow + r;
    float sc = si * sW[m];
    Ob[(size_t)m * N + n0]      = (float)c0[r] * sc;
    Ob[(size_t)m * N + n0 + 16] = (float)c1[r] * sc;
  }
}

__global__ void k_dw5x5(const int8_t* __restrict__ qin,
                        const int8_t* __restrict__ qW,
                        const float* __restrict__ sWdw,
                        const float* __restrict__ s_cat,
                        const float* __restrict__ s_pw,
                        int8_t* __restrict__ qout) {
  size_t idx = (size_t)blockIdx.x * 256 + threadIdx.x;
  int n = idx & (HW - 1);
  int c = (idx >> 10) & (TOTAL - 1);
  int yy = n >> 5, xx = n & 31;
  const int8_t* plane = qin + (idx - n);
  int acc = 0;
  for (int dy = -2; dy <= 2; ++dy) {
    int y2 = yy + dy;
    if (y2 < 0 || y2 > 31) continue;
    for (int dx = -2; dx <= 2; ++dx) {
      int x2 = xx + dx;
      if (x2 < 0 || x2 > 31) continue;
      acc += (int)plane[y2 * 32 + x2] * (int)qW[c * 25 + (dy + 2) * 5 + (dx + 2)];
    }
  }
  float v = (float)acc * (*s_cat) * sWdw[c];
  qout[idx] = (int8_t)(int)fq8(v, *s_pw);
}

__global__ void k_grouped_pw(const int8_t* __restrict__ qx1,
                             const int8_t* __restrict__ qWpw,
                             const float* __restrict__ sWpw,
                             const float* __restrict__ s_pw,
                             const float* __restrict__ s_cat,
                             int8_t* __restrict__ qcat2) {
  size_t idx = (size_t)blockIdx.x * 256 + threadIdx.x;
  int n = idx & (HW - 1);
  int o = (idx >> 10) & (TOTAL - 1);
  int b = (int)(idx >> 18);
  int g = o >> 3;
  const int8_t* in = qx1 + ((size_t)b * TOTAL + g * 8) * HW + n;
  int acc = 0;
  for (int e = 0; e < 8; ++e)
    acc += (int)in[(size_t)e * HW] * (int)qWpw[o * 8 + e];
  float v = (float)acc * (*s_pw) * sWpw[o];
  qcat2[idx] = (int8_t)(int)fq8(v, *s_cat);
}

#if USE_TDM
__device__ __forceinline__ void tdm_load_1d(unsigned lds_off, const void* g,
                                            unsigned bytes) {
  unsigned long long ga = (unsigned long long)(uintptr_t)g;
  v4u g0;
  g0[0] = 1u;
  g0[1] = lds_off;
  g0[2] = (unsigned)ga;
  g0[3] = (unsigned)((ga >> 32) & 0x01FFFFFFull) | (2u << 30);
  v8i g1;
  g1[0] = 0;
  g1[1] = (int)((bytes & 0xFFFFu) << 16);
  g1[2] = (int)(((bytes >> 16) & 0xFFFFu) | (1u << 16));
  g1[3] = (int)((bytes & 0xFFFFu) << 16);
  g1[4] = 1;
  g1[5] = (int)bytes;
  g1[6] = 0; g1[7] = 0;
  v4i z4 = {0, 0, 0, 0};
  v8i z8 = {0, 0, 0, 0, 0, 0, 0, 0};
  __builtin_amdgcn_tensor_load_to_lds(g0, g1, z4, z4, z8, 0);
}
#endif

#define SM_Q    0
#define SM_K    8192
#define SM_V    16384
#define SM_RED  24576
#define SM_REDK (24576 + 1024)
#define SM_KVF  (24576 + 1152)
#define SM_KSUM (24576 + 1408)
#define SM_BYTES (24576 + 1440)

__global__ void k_attention(const int8_t* __restrict__ qxq_q,
                            const int8_t* __restrict__ qcat2_q,
                            const int8_t* __restrict__ qxq_k,
                            const int8_t* __restrict__ qcat2_k,
                            const int8_t* __restrict__ qxq_v,
                            const int8_t* __restrict__ qcat2_v,
                            const float* __restrict__ s_cat_q,
                            const float* __restrict__ s_cat_k,
                            const float* __restrict__ s_cat_v,
                            const float* __restrict__ s_kv,
                            const float* __restrict__ s_qkv,
                            const float* __restrict__ s_proj_in,
                            int8_t* __restrict__ qout) {
  extern __shared__ char smem[];
  int8_t* qs = (int8_t*)(smem + SM_Q);
  int8_t* ks = (int8_t*)(smem + SM_K);
  int8_t* vs = (int8_t*)(smem + SM_V);
  int (*red)[4]  = (int(*)[4])(smem + SM_RED);
  int (*redk)[4] = (int(*)[4])(smem + SM_REDK);
  float* kvf   = (float*)(smem + SM_KVF);
  float* ksumf = (float*)(smem + SM_KSUM);

  int b = blockIdx.x >> 6;
  int g = blockIdx.x & 63;
  bool hi = (g >= HEADS);
  size_t base = ((size_t)b * TOTAL + (hi ? (g - HEADS) : g) * DIM) * HW;
  const int8_t* sq = (hi ? qcat2_q : qxq_q) + base;
  const int8_t* sk = (hi ? qcat2_k : qxq_k) + base;
  const int8_t* sv = (hi ? qcat2_v : qxq_v) + base;

  int t = threadIdx.x;
#if USE_TDM
  if (t < 32) {
    unsigned lds0 = (unsigned)__builtin_amdgcn_groupstaticsize();
    tdm_load_1d(lds0 + SM_Q, sq, DIM * HW);
    tdm_load_1d(lds0 + SM_K, sk, DIM * HW);
    tdm_load_1d(lds0 + SM_V, sv, DIM * HW);
    __builtin_amdgcn_s_wait_tensorcnt(0);
  }
#else
  for (int i = t; i < DIM * HW; i += 256) {
    qs[i] = sq[i];
    ks[i] = sk[i];
    vs[i] = sv[i];
  }
#endif
  __syncthreads();

  {
    int p = t & 63, qr = t >> 6;
    int d = p >> 3, e = p & 7;
    int acc = 0;
    int n0 = qr * 256;
    for (int n = n0; n < n0 + 256; ++n) {
      int kd = ks[d * HW + n];
      kd = kd > 0 ? kd : 0;
      acc += kd * (int)vs[e * HW + n];
    }
    red[p][qr] = acc;
  }
  __syncthreads();

  float sck = *s_cat_k, scv = *s_cat_v, skv = *s_kv;
  if (t < 64) {
    int tot = red[t][0] + red[t][1] + red[t][2] + red[t][3];
    kvf[t] = fq8((float)tot * sck * scv, skv) * skv;
  }
  if (t >= 64 && t < 96) {
    int d = (t - 64) >> 2, qr = (t - 64) & 3;
    int acc = 0;
    int n0 = qr * 256;
    for (int n = n0; n < n0 + 256; ++n) {
      int kd = ks[d * HW + n];
      acc += (kd > 0 ? kd : 0);
    }
    redk[d][qr] = acc;
  }
  __syncthreads();
  if (t < DIM)
    ksumf[t] = (float)(redk[t][0] + redk[t][1] + redk[t][2] + redk[t][3]) * sck;
  __syncthreads();

  float scq = *s_cat_q, sqk = *s_qkv, spin = *s_proj_in;
  for (int n = t; n < HW; n += 256) {
    float qf[DIM];
    float denom = 0.f;
    for (int d = 0; d < DIM; ++d) {
      int qv = qs[d * HW + n];
      qv = qv > 0 ? qv : 0;
      qf[d] = (float)qv * scq;
      denom += qf[d] * ksumf[d];
    }
    denom += 1e-15f;
    size_t ob = ((size_t)b * HW + n) * C2 + (size_t)g * DIM;
    for (int e = 0; e < DIM; ++e) {
      float acc = 0.f;
      for (int d = 0; d < DIM; ++d) acc += qf[d] * kvf[d * 8 + e];
      float qkvq = fq8(acc, sqk) * sqk;
      float o = qkvq / denom;
      qout[ob + e] = (int8_t)(int)fq8(o, spin);
    }
  }
}

__global__ void k_bn_stats(const float* __restrict__ y, float* __restrict__ stats) {
  __shared__ float ssum[256], ssq[256];
  int m = blockIdx.x;
  float s = 0.f, s2 = 0.f;
  for (int i = threadIdx.x; i < BATCH * HW; i += 256) {
    int b = i >> 10, n = i & (HW - 1);
    float v = y[((size_t)b * CIN + m) * HW + n];
    s += v; s2 += v * v;
  }
  ssum[threadIdx.x] = s; ssq[threadIdx.x] = s2;
  __syncthreads();
  for (int o = 128; o > 0; o >>= 1) {
    if (threadIdx.x < (unsigned)o) {
      ssum[threadIdx.x] += ssum[threadIdx.x + o];
      ssq[threadIdx.x]  += ssq[threadIdx.x + o];
    }
    __syncthreads();
  }
  if (threadIdx.x == 0) {
    float inv = 1.f / (float)(BATCH * HW);
    float mean = ssum[0] * inv;
    stats[m * 2 + 0] = mean;
    stats[m * 2 + 1] = ssq[0] * inv - mean * mean;
  }
}

__global__ void k_final(const float* __restrict__ y,
                        const float* __restrict__ stats,
                        const float* __restrict__ gamma,
                        const float* __restrict__ beta,
                        const int8_t* __restrict__ qx,
                        const float* __restrict__ s_in,
                        const float* __restrict__ s_out,
                        float* __restrict__ out) {
  size_t idx = (size_t)blockIdx.x * 256 + threadIdx.x;
  int n = idx & (HW - 1);
  int c = (idx >> 10) & (CIN - 1);
  int b = (int)(idx >> 18);
  float so = *s_out;
  float mean = stats[c * 2 + 0], var = stats[c * 2 + 1];
  float norm = gamma[c] * (y[idx] - mean) * rsqrtf(var + 1e-5f) + beta[c];
  float f1 = rintf(fminf(fmaxf(norm / so, -32768.f), 32767.f)) * so;
  float xr = (float)qx[((size_t)b * HW + n) * CIN + c] * (*s_in);
  float f2 = rintf(fminf(fmaxf(xr / so, -32768.f), 32767.f)) * so;
  out[idx] = f1 + f2;
}

extern "C" void kernel_launch(void* const* d_in, const int* in_sizes, int n_in,
                              void* d_out, int out_size, void* d_ws, size_t ws_size,
                              hipStream_t stream) {
  (void)in_sizes; (void)n_in; (void)out_size; (void)ws_size;

  const float* x     = (const float*)d_in[0];
  const float* s_in  = (const float*)d_in[1];
  const float* Wt[3]   = {(const float*)d_in[2], (const float*)d_in[4], (const float*)d_in[6]};
  const float* sWt[3]  = {(const float*)d_in[3], (const float*)d_in[5], (const float*)d_in[7]};
  const float* Wdw[3]  = {(const float*)d_in[8],  (const float*)d_in[14], (const float*)d_in[20]};
  const float* sWdw[3] = {(const float*)d_in[9],  (const float*)d_in[15], (const float*)d_in[21]};
  const float* Wpw[3]  = {(const float*)d_in[10], (const float*)d_in[16], (const float*)d_in[22]};
  const float* sWpw[3] = {(const float*)d_in[11], (const float*)d_in[17], (const float*)d_in[23]};
  const float* s_pw[3]  = {(const float*)d_in[12], (const float*)d_in[18], (const float*)d_in[24]};
  const float* s_cat[3] = {(const float*)d_in[13], (const float*)d_in[19], (const float*)d_in[25]};
  const float* s_kv      = (const float*)d_in[26];
  const float* s_qkv     = (const float*)d_in[27];
  const float* Wproj     = (const float*)d_in[28];
  const float* sWproj    = (const float*)d_in[29];
  const float* s_proj_in = (const float*)d_in[30];
  const float* bn_gamma  = (const float*)d_in[31];
  const float* bn_beta   = (const float*)d_in[32];
  const float* s_out     = (const float*)d_in[33];
  float* out = (float*)d_out;

  char* w = (char*)d_ws;
  size_t off = 0;
  auto alloc = [&](size_t bytes) -> void* {
    off = (off + 255) & ~(size_t)255;
    void* p = w + off;
    off += bytes;
    return p;
  };
  const size_t NBHWC = (size_t)BATCH * HW * CIN;
  int8_t* qx      = (int8_t*)alloc(NBHWC);
  int8_t* qWt[3]; for (int i = 0; i < 3; ++i) qWt[i]  = (int8_t*)alloc(TOTAL * CIN);
  int8_t* qWdw[3]; for (int i = 0; i < 3; ++i) qWdw[i] = (int8_t*)alloc(TOTAL * 25);
  int8_t* qWpw[3]; for (int i = 0; i < 3; ++i) qWpw[i] = (int8_t*)alloc(TOTAL * DIM);
  int8_t* qWproj  = (int8_t*)alloc((size_t)CIN * C2);
  int8_t* qxq[3];  for (int i = 0; i < 3; ++i) qxq[i]   = (int8_t*)alloc(NBHWC);
  int8_t* qx1     = (int8_t*)alloc(NBHWC);
  int8_t* qcat2[3]; for (int i = 0; i < 3; ++i) qcat2[i] = (int8_t*)alloc(NBHWC);
  int8_t* qout    = (int8_t*)alloc((size_t)BATCH * HW * C2);
  float*  y       = (float*)alloc(NBHWC * sizeof(float));
  float*  stats   = (float*)alloc(CIN * 2 * sizeof(float));

  const dim3 blk(256);

  k_quant_input<<<dim3((unsigned)(NBHWC / 256)), blk, 0, stream>>>(x, s_in, qx);

  for (int i = 0; i < 3; ++i) {
    k_quant_weight<<<dim3((TOTAL * CIN + 255) / 256), blk, 0, stream>>>(Wt[i], sWt[i], qWt[i], CIN, TOTAL * CIN);
    k_quant_weight<<<dim3((TOTAL * 25 + 255) / 256), blk, 0, stream>>>(Wdw[i], sWdw[i], qWdw[i], 25, TOTAL * 25);
    k_quant_weight<<<dim3((TOTAL * DIM + 255) / 256), blk, 0, stream>>>(Wpw[i], sWpw[i], qWpw[i], DIM, TOTAL * DIM);
  }
  k_quant_weight<<<dim3((CIN * C2 + 255) / 256), blk, 0, stream>>>(Wproj, sWproj, qWproj, C2, CIN * C2);

  const dim3 ggrid(TOTAL / 32, HW / 128, BATCH);
  for (int i = 0; i < 3; ++i) {
    k_gemm_iu8_quant<<<ggrid, blk, 0, stream>>>(qWt[i], qx, qxq[i], sWt[i], s_in, s_cat[i],
                                                TOTAL, HW, CIN);
    k_dw5x5<<<dim3((unsigned)(NBHWC / 256)), blk, 0, stream>>>(qxq[i], qWdw[i], sWdw[i],
                                                               s_cat[i], s_pw[i], qx1);
    k_grouped_pw<<<dim3((unsigned)(NBHWC / 256)), blk, 0, stream>>>(qx1, qWpw[i], sWpw[i],
                                                                    s_pw[i], s_cat[i], qcat2[i]);
  }

  k_attention<<<dim3(BATCH * H2), blk, SM_BYTES, stream>>>(
      qxq[0], qcat2[0], qxq[1], qcat2[1], qxq[2], qcat2[2],
      s_cat[0], s_cat[1], s_cat[2], s_kv, s_qkv, s_proj_in, qout);

  k_gemm_iu8_f32<<<ggrid, blk, 0, stream>>>(qWproj, qout, y, sWproj, s_proj_in,
                                            CIN, HW, C2);

  k_bn_stats<<<dim3(CIN), blk, 0, stream>>>(y, stats);

  k_final<<<dim3((unsigned)(NBHWC / 256)), blk, 0, stream>>>(y, stats, bn_gamma, bn_beta,
                                                             qx, s_in, s_out, out);
}
